// HAB_D_15642270892594
// MI455X (gfx1250) — matrix operand order repaired and checked by reading, not compiled, not run on hardware
//
#include <hip/hip_runtime.h>

// ---------------------------------------------------------------------------
// Types for CDNA5 WMMA (gfx1250, wave32)
// ---------------------------------------------------------------------------
typedef __bf16 bf16;
typedef __attribute__((ext_vector_type(8)))  bf16  v8bf;
typedef __attribute__((ext_vector_type(16))) bf16  v16bf;
typedef __attribute__((ext_vector_type(8)))  float v8f;

union FragU { v16bf v; v8bf h[2]; };

__device__ __forceinline__ v8f wmma_bf16(v16bf a, v16bf b, v8f c) {
  return __builtin_amdgcn_wmma_f32_16x16x32_bf16(
      /*neg_a=*/false, a, /*neg_b=*/false, b,
      /*c_mod=*/(short)0, c, /*reuse_a=*/false, /*reuse_b=*/false);
}

// CDNA5 async DMA: global -> LDS, 16 bytes per lane, tracked by ASYNCcnt.
// GV mode: 64-bit global address in a VGPR pair, LDS byte offset in a VGPR
// (low 32 bits of the generic shared pointer are the LDS offset per the
// aperture rules in CDNA5 ISA).
__device__ __forceinline__ void async_b128(void* lds, const void* gptr) {
  unsigned l = (unsigned)(uintptr_t)lds;
  unsigned long long g = (unsigned long long)(uintptr_t)gptr;
  asm volatile("global_load_async_to_lds_b128 %0, %1, off"
               :: "v"(l), "v"(g) : "memory");
}
__device__ __forceinline__ void wait_async0() {
  asm volatile("s_wait_asynccnt 0" ::: "memory");
}

// A-matrix fragment (16xK tile, row-major, 16-bit): lane 0-15 -> row m, K halves
// {0..7,16..23}; lane 16-31 -> {8..15,24..31}. base points at (row0, k0).
__device__ __forceinline__ v16bf frag_a(const bf16* base, int stride) {
  int lane = threadIdx.x & 31;
  int m = lane & 15, hl = lane >> 4;
  const bf16* p = base + m * stride + hl * 8;
  FragU u;
  u.h[0] = *(const v8bf*)p;
  u.h[1] = *(const v8bf*)(p + 16);
  return u.v;
}

// B-matrix fragment read from a transposed [n][k] array: lane 0-15 hold K
// {0..7,16..23}, lane 16-31 hold K {8..15,24..31} (matches frag_a).
__device__ __forceinline__ v16bf frag_b(const bf16* base, int stride) {
  int lane = threadIdx.x & 31;
  int n = lane & 15, hl = lane >> 4;
  const bf16* p = base + n * stride + hl * 8;
  FragU u;
  u.h[0] = *(const v8bf*)p;
  u.h[1] = *(const v8bf*)(p + 16);
  return u.v;
}

__device__ __forceinline__ float wred_sum(float v) {
  #pragma unroll
  for (int i = 16; i > 0; i >>= 1) v += __shfl_xor(v, i, 32);
  return v;
}
__device__ __forceinline__ float wred_max(float v) {
  #pragma unroll
  for (int i = 16; i > 0; i >>= 1) v = fmaxf(v, __shfl_xor(v, i, 32));
  return v;
}

__device__ __forceinline__ float gelu_f(float x) {
  return 0.5f * x * (1.f + erff(x * 0.70710678118f));
}

// spatial token (b,h,w) -> window-ordered row
__device__ __forceinline__ int winrow(int b, int h, int w) {
  return ((((b << 3) + (h >> 4)) << 3) + (w >> 4)) * 256 + ((h & 15) << 4) + (w & 15);
}

// ---------------------------------------------------------------------------
// Weight pack: f32 [taps*Cin][N] (row-major over N) -> bf16 Bt [Np][taps*Cpad]
// ---------------------------------------------------------------------------
__global__ void pack_w(const float* __restrict__ w, bf16* __restrict__ out,
                       int N, int Np, int Cin, int Cpad, int taps) {
  int Kp = taps * Cpad;
  long long idx = (long long)blockIdx.x * 256 + threadIdx.x;
  if (idx >= (long long)Np * Kp) return;
  int n = (int)(idx / Kp), k = (int)(idx % Kp);
  int tap = k / Cpad, ci = k - tap * Cpad;
  float v = 0.f;
  if (n < N && ci < Cin) v = w[(long long)(tap * Cin + ci) * N + n];
  out[idx] = (bf16)v;
}

// ---------------------------------------------------------------------------
// LayerNorm over C=180 (one wave per token), bf16 out padded to 192 cols,
// optionally written in window-partitioned row order.
// ---------------------------------------------------------------------------
__global__ __launch_bounds__(256) void ln_kernel(
    const float* __restrict__ x, const float* __restrict__ g,
    const float* __restrict__ beta, bf16* __restrict__ out, int windowize) {
  int lane = threadIdx.x & 31, wave = threadIdx.x >> 5;
  int tok = blockIdx.x * 8 + wave;
  const float* row = x + (size_t)tok * 180;
  float vals[6], s = 0.f, sq = 0.f;
  #pragma unroll
  for (int i = 0; i < 6; ++i) {
    int c = lane + i * 32;
    float v = (c < 180) ? row[c] : 0.f;
    vals[i] = v; s += v; sq += v * v;
  }
  s = wred_sum(s); sq = wred_sum(sq);
  float mu = s * (1.f / 180.f);
  float var = sq * (1.f / 180.f) - mu * mu;
  float rstd = rsqrtf(var + 1e-5f);
  int orow = tok;
  if (windowize) {
    int b = tok >> 14, h = (tok >> 7) & 127, w = tok & 127;
    orow = winrow(b, h, w);
  }
  bf16* op = out + (size_t)orow * 192;
  #pragma unroll
  for (int i = 0; i < 6; ++i) {
    int c = lane + i * 32;
    float v = (c < 180) ? (vals[i] - mu) * rstd * g[c] + beta[c] : 0.f;
    op[c] = (bf16)v;
  }
}

// ---------------------------------------------------------------------------
// WMMA GEMM: out[M x N] = A[M x Kp] * Bt^T (+bias)(+res)(act), M mult of 128.
// Block tile 128x64, BK=32, 8 waves -> 32x32 each (2x2 WMMA fragments).
// Plain path: double-buffered GLOBAL_LOAD_ASYNC_TO_LDS pipeline (ASYNCcnt),
// one barrier per K-step, async DMA overlaps WMMA issue.
// IM2COL path: predicated 3x3 gather (needs zero-fill, stays on ds_store)
// with global_prefetch of the next weight tile.
// ---------------------------------------------------------------------------
template <bool IM2COL, bool WININ>
__global__ __launch_bounds__(256) void gemm_kernel(
    const bf16* __restrict__ Aptr, const bf16* __restrict__ Bt,
    const float* __restrict__ bias, const float* __restrict__ res,
    float* __restrict__ outf, bf16* __restrict__ outb,
    int Nreal, int Npad, int Kp, int Cpad, int act) {
  constexpr int NBUF = IM2COL ? 1 : 2;
  __shared__ bf16 As[NBUF][128 * 32];
  __shared__ bf16 Bs[NBUF][64 * 32];
  int tid = threadIdx.x, lane = tid & 31, wave = tid >> 5;
  int m0 = blockIdx.x * 128, n0 = blockIdx.y * 64;
  int wm = wave >> 1, wn = wave & 1;
  int ar = tid >> 1, aseg = (tid & 1) * 16;
  int bn = tid >> 2, bseg = (tid & 3) * 8;

  v8f acc00 = {}, acc01 = {}, acc10 = {}, acc11 = {};
  int nk = Kp / 32;

  if constexpr (!IM2COL) {
    // ---- async double-buffered pipeline ----
    {  // prologue: tile 0 -> buffer 0
      const bf16* pa = Aptr + (size_t)(m0 + ar) * Kp + aseg;
      bf16* la = &As[0][ar * 32 + aseg];
      async_b128(la, pa);
      async_b128(la + 8, pa + 8);
      async_b128(&Bs[0][bn * 32 + bseg], Bt + (size_t)(n0 + bn) * Kp + bseg);
    }
    for (int kt = 0; kt < nk; ++kt) {
      wait_async0();       // this wave's tile-kt data is in LDS
      __syncthreads();     // ...and everyone else's too
      if (kt + 1 < nk) {   // issue tile kt+1 into the other buffer
        int k0 = (kt + 1) * 32, b = (kt + 1) & 1;
        const bf16* pa = Aptr + (size_t)(m0 + ar) * Kp + k0 + aseg;
        bf16* la = &As[b][ar * 32 + aseg];
        async_b128(la, pa);
        async_b128(la + 8, pa + 8);
        async_b128(&Bs[b][bn * 32 + bseg],
                   Bt + (size_t)(n0 + bn) * Kp + k0 + bseg);
      }
      const bf16* Ac = As[kt & 1];
      const bf16* Bc = Bs[kt & 1];
      v16bf a0 = frag_a(Ac + (wm * 32) * 32, 32);
      v16bf a1 = frag_a(Ac + (wm * 32 + 16) * 32, 32);
      v16bf b0 = frag_b(Bc + (wn * 32) * 32, 32);
      v16bf b1 = frag_b(Bc + (wn * 32 + 16) * 32, 32);
      acc00 = wmma_bf16(a0, b0, acc00);
      acc01 = wmma_bf16(a0, b1, acc01);
      acc10 = wmma_bf16(a1, b0, acc10);
      acc11 = wmma_bf16(a1, b1, acc11);
      // no trailing barrier: next iteration's barrier protects buffer reuse
    }
  } else {
    // ---- synchronous im2col path ----
    for (int kt = 0; kt < nk; ++kt) {
      int k0 = kt * 32;
      {  // A tile (gathered 3x3 neighborhood, zero-filled out of bounds)
        int m = m0 + ar, k = k0 + aseg;
        v8bf lo = {}, hi = {};
        int tap = k / Cpad, ci = k - tap * Cpad;
        int dy = tap / 3 - 1, dx = tap % 3 - 1;
        int bb = m >> 14, hh = (m >> 7) & 127, ww = m & 127;
        int nh = hh + dy, nw = ww + dx;
        if (nh >= 0 && nh < 128 && nw >= 0 && nw < 128) {
          int row = WININ ? winrow(bb, nh, nw) : ((bb << 14) | (nh << 7) | nw);
          const bf16* p = Aptr + (size_t)row * Cpad + ci;
          lo = *(const v8bf*)p; hi = *(const v8bf*)(p + 8);
        }
        *(v8bf*)(&As[0][ar * 32 + aseg]) = lo;
        *(v8bf*)(&As[0][ar * 32 + aseg + 8]) = hi;
      }
      *(v8bf*)(&Bs[0][bn * 32 + bseg]) =
          *(const v8bf*)(Bt + (size_t)(n0 + bn) * Kp + k0 + bseg);
      if (k0 + 32 < Kp)  // -> global_prefetch_b8 for the next weight tile
        __builtin_prefetch(Bt + (size_t)(n0 + bn) * Kp + k0 + 32 + bseg, 0, 1);
      __syncthreads();

      v16bf a0 = frag_a(&As[0][(wm * 32) * 32], 32);
      v16bf a1 = frag_a(&As[0][(wm * 32 + 16) * 32], 32);
      v16bf b0 = frag_b(&Bs[0][(wn * 32) * 32], 32);
      v16bf b1 = frag_b(&Bs[0][(wn * 32 + 16) * 32], 32);
      acc00 = wmma_bf16(a0, b0, acc00);
      acc01 = wmma_bf16(a0, b1, acc01);
      acc10 = wmma_bf16(a1, b0, acc10);
      acc11 = wmma_bf16(a1, b1, acc11);
      __syncthreads();
    }
  }

  // ---- epilogue ----
  int hl = lane >> 4;
  v8f accs[2][2] = {{acc00, acc01}, {acc10, acc11}};
  #pragma unroll
  for (int ii = 0; ii < 2; ++ii)
    #pragma unroll
    for (int jj = 0; jj < 2; ++jj)
      #pragma unroll
      for (int i = 0; i < 8; ++i) {
        int m = m0 + wm * 32 + ii * 16 + i + hl * 8;
        int col = n0 + wn * 32 + jj * 16 + (lane & 15);
        float v = accs[ii][jj][i];
        if (col < Nreal) {
          if (bias) v += bias[col];
          if (res) v += res[(size_t)m * Nreal + col];
          if (act == 1) v = gelu_f(v);
          if (outf) outf[(size_t)m * Nreal + col] = v;
          if (outb) outb[(size_t)m * Npad + col] = (bf16)v;
        } else if (outb) {
          outb[(size_t)m * Npad + col] = (bf16)0.f;
        }
      }
}

// ---------------------------------------------------------------------------
// Repack qkv (bf16, stride 576, window-ordered rows) into per-head padded
// Q,K (rows [win*6+h]*256+p, 32 cols) and V transposed ([win*6+h]*32+d, 256).
// Q pre-scaled by HD^-0.5.
// ---------------------------------------------------------------------------
__global__ void repack_qkv(const bf16* __restrict__ qkv, bf16* __restrict__ Q,
                           bf16* __restrict__ K, bf16* __restrict__ V) {
  int idx = blockIdx.x * 256 + threadIdx.x;  // 65536*192
  int row = idx / 192, r2 = idx - row * 192;
  int h = r2 >> 5, d = r2 & 31;
  int win = row >> 8, p = row & 255;
  float q = 0.f, kk = 0.f, vv = 0.f;
  if (d < 30) {
    int c = h * 30 + d;
    const bf16* src = qkv + (size_t)row * 576;
    q  = (float)src[c] * 0.18257418583506f;  // 30^-0.5
    kk = (float)src[180 + c];
    vv = (float)src[360 + c];
  }
  int qi = ((win * 6 + h) * 256 + p) * 32 + d;
  Q[qi] = (bf16)q;
  K[qi] = (bf16)kk;
  V[((win * 6 + h) * 32 + d) * 256 + p] = (bf16)vv;
}

// ---------------------------------------------------------------------------
// Fused window attention: block = (16-row slab rb, head, window), 256 thr.
// Q/K/V staged via GLOBAL_LOAD_ASYNC_TO_LDS. S = Q K^T (WMMA) -> +bias ->
// softmax (wave32 shuffles) -> P (bf16, in-place alias) -> O = P V (WMMA,
// K split over waves + LDS reduce).
// ---------------------------------------------------------------------------
__global__ __launch_bounds__(256) void attn_kernel(
    const bf16* __restrict__ Qb, const bf16* __restrict__ Kb,
    const bf16* __restrict__ Vb, const int* __restrict__ rpi,
    const float* __restrict__ rpb, bf16* __restrict__ O) {
  int rb = blockIdx.x, head = blockIdx.y, win = blockIdx.z;
  int tid = threadIdx.x, lane = tid & 31, wave = tid >> 5;

  __shared__ bf16 Qs[16 * 32];
  __shared__ bf16 Ks[256 * 32];
  __shared__ bf16 Vt[32 * 256];
  __shared__ float Ss[16 * 256];          // 16 KB; aliased below
  bf16* Pb = (bf16*)Ss;                   // bytes [0, 8K): bf16 probabilities
  float* Opart = Ss + 2048;               // bytes [8K,16K): partial O tiles

  int qkbase = (win * 6 + head) * 256;
  {  // K: one 32-bf16 row per thread (4x async b128)
    const bf16* s = Kb + (size_t)(qkbase + tid) * 32;
    bf16* d = Ks + tid * 32;
    async_b128(d, s);       async_b128(d + 8, s + 8);
    async_b128(d + 16, s + 16); async_b128(d + 24, s + 24);
  }
  if (tid < 64) {  // Q slab: 16 rows x 32
    int r = tid >> 2, seg = (tid & 3) * 8;
    async_b128(Qs + r * 32 + seg,
               Qb + (size_t)(qkbase + rb * 16 + r) * 32 + seg);
  }
  {  // V^T: 32 rows x 256
    int d = tid >> 3, c0 = (tid & 7) * 32;
    const bf16* s = Vb + (size_t)((win * 6 + head) * 32 + d) * 256 + c0;
    bf16* dv = Vt + d * 256 + c0;
    async_b128(dv, s);       async_b128(dv + 8, s + 8);
    async_b128(dv + 16, s + 16); async_b128(dv + 24, s + 24);
  }
  wait_async0();
  __syncthreads();

  // ---- S = Q K^T : 16 column tiles, 2 per wave ----
  v16bf aq = frag_a(Qs, 32);
  #pragma unroll
  for (int j = 0; j < 2; ++j) {
    int nt = wave * 2 + j;
    v16bf bk = frag_b(Ks + nt * 16 * 32, 32);
    v8f acc = {};
    acc = wmma_bf16(aq, bk, acc);
    int hl = lane >> 4;
    #pragma unroll
    for (int i = 0; i < 8; ++i)
      Ss[(i + hl * 8) * 256 + nt * 16 + (lane & 15)] = acc[i];
  }
  __syncthreads();

  // ---- softmax (+bias): 2 rows per wave, 8 cols per lane ----
  float pv[2][8];
  #pragma unroll
  for (int rr = 0; rr < 2; ++rr) {
    int r = wave * 2 + rr, rg = rb * 16 + r;
    float t[8], mx = -1e30f;
    #pragma unroll
    for (int u = 0; u < 8; ++u) {
      int c = lane + u * 32;
      int bi = rpi[rg * 256 + c];
      t[u] = Ss[r * 256 + c] + rpb[bi * 6 + head];
      mx = fmaxf(mx, t[u]);
    }
    mx = wred_max(mx);
    float sum = 0.f;
    #pragma unroll
    for (int u = 0; u < 8; ++u) { t[u] = __expf(t[u] - mx); sum += t[u]; }
    sum = wred_sum(sum);
    float inv = 1.f / sum;
    #pragma unroll
    for (int u = 0; u < 8; ++u) pv[rr][u] = t[u] * inv;
  }
  __syncthreads();  // all reads of f32 S done before bf16 alias writes
  #pragma unroll
  for (int rr = 0; rr < 2; ++rr) {
    int r = wave * 2 + rr;
    #pragma unroll
    for (int u = 0; u < 8; ++u)
      Pb[r * 256 + lane + u * 32] = (bf16)pv[rr][u];
  }
  __syncthreads();

  // ---- O = P V : 2 col tiles x 4 K-quarters across 8 waves ----
  {
    int nt = wave & 1, kq = wave >> 1;
    v8f acc = {};
    #pragma unroll
    for (int s2 = 0; s2 < 2; ++s2) {
      int kb = kq * 64 + s2 * 32;
      v16bf ap = frag_a(Pb + kb, 256);
      v16bf bv = frag_b(Vt + nt * 16 * 256 + kb, 256);
      acc = wmma_bf16(ap, bv, acc);
    }
    int hl = lane >> 4;
    #pragma unroll
    for (int i = 0; i < 8; ++i)
      Opart[((kq * 2 + nt) * 16 + (i + hl * 8)) * 16 + (lane & 15)] = acc[i];
  }
  __syncthreads();
  if (wave < 2) {
    int nt = wave, hl = lane >> 4;
    #pragma unroll
    for (int i = 0; i < 8; ++i) {
      int r = i + hl * 8, c16 = lane & 15;
      float s = 0.f;
      #pragma unroll
      for (int kq = 0; kq < 4; ++kq)
        s += Opart[((kq * 2 + nt) * 16 + r) * 16 + c16];
      int d = nt * 16 + c16;
      if (d < 30) {
        int rowg = win * 256 + rb * 16 + r;
        O[(size_t)rowg * 192 + head * 30 + d] = (bf16)s;
      }
    }
  }
}

// ---------------------------------------------------------------------------
// Channel-attention helpers and residual combine
// ---------------------------------------------------------------------------
__global__ __launch_bounds__(256) void pool_kernel(const float* __restrict__ y,
                                                   float* __restrict__ pooled) {
  int bc = blockIdx.x;  // 0..719
  int b = bc / 180, c = bc - b * 180;
  float s = 0.f;
  for (int i = threadIdx.x; i < 16384; i += 256)
    s += y[((size_t)b * 16384 + i) * 180 + c];
  __shared__ float red[256];
  red[threadIdx.x] = s; __syncthreads();
  for (int st = 128; st > 0; st >>= 1) {
    if (threadIdx.x < st) red[threadIdx.x] += red[threadIdx.x + st];
    __syncthreads();
  }
  if (threadIdx.x == 0) pooled[bc] = red[0] * (1.f / 16384.f);
}

__global__ __launch_bounds__(256) void ca_kernel(
    const float* __restrict__ pooled, const float* __restrict__ w1,
    const float* __restrict__ b1, const float* __restrict__ w2,
    const float* __restrict__ b2, float* __restrict__ a) {
  __shared__ float hid[24];
  int t = threadIdx.x;
  if (t < 24) {
    int b = t / 6, j = t - b * 6;
    float s = b1[j];
    for (int c = 0; c < 180; ++c) s += pooled[b * 180 + c] * w1[c * 6 + j];
    hid[t] = fmaxf(s, 0.f);
  }
  __syncthreads();
  for (int idx = t; idx < 720; idx += 256) {
    int b = idx / 180, c = idx - b * 180;
    float s = b2[c];
    #pragma unroll
    for (int j = 0; j < 6; ++j) s += hid[b * 6 + j] * w2[j * 180 + c];
    a[idx] = 1.f / (1.f + __expf(-s));
  }
}

__global__ void combine_kernel(const float* __restrict__ x,
                               const float* __restrict__ attn,
                               const float* __restrict__ y2,
                               const float* __restrict__ a,
                               float* __restrict__ xres) {
  int idx = blockIdx.x * 256 + threadIdx.x;
  if (idx >= 65536 * 180) return;
  int m = idx / 180, c = idx - m * 180;
  int b = m >> 14, h = (m >> 7) & 127, w = m & 127;
  int wr = winrow(b, h, w);
  xres[idx] = x[idx] + attn[(size_t)wr * 180 + c] + y2[idx] * a[b * 180 + c] * 0.01f;
}

// ---------------------------------------------------------------------------
// Host launcher
// ---------------------------------------------------------------------------
extern "C" void kernel_launch(void* const* d_in, const int* in_sizes, int n_in,
                              void* d_out, int out_size, void* d_ws, size_t ws_size,
                              hipStream_t stream) {
  (void)in_sizes; (void)n_in; (void)out_size; (void)ws_size;
  const float* x    = (const float*)d_in[0];
  const int*   rpi  = (const int*)d_in[1];
  const float* n1g  = (const float*)d_in[6];
  const float* n1b  = (const float*)d_in[7];
  const float* qkvw = (const float*)d_in[8];
  const float* qkvb = (const float*)d_in[9];
  const float* rpb  = (const float*)d_in[10];
  const float* prw  = (const float*)d_in[11];
  const float* prb  = (const float*)d_in[12];
  const float* c1w  = (const float*)d_in[13];
  const float* c1b  = (const float*)d_in[14];
  const float* c2w  = (const float*)d_in[15];
  const float* c2b  = (const float*)d_in[16];
  const float* ca1w = (const float*)d_in[17];
  const float* ca1b = (const float*)d_in[18];
  const float* ca2w = (const float*)d_in[19];
  const float* ca2b = (const float*)d_in[20];
  const float* n2g  = (const float*)d_in[21];
  const float* n2b  = (const float*)d_in[22];
  const float* f1w  = (const float*)d_in[23];
  const float* f1b  = (const float*)d_in[24];
  const float* f2w  = (const float*)d_in[25];
  const float* f2b  = (const float*)d_in[26];
  float* out = (float*)d_out;

  char* base = (char*)d_ws;
  size_t off = 0;
  auto alloc = [&](size_t bytes) -> void* {
    size_t o = (off + 255) & ~(size_t)255;
    off = o + bytes;
    return (void*)(base + o);
  };

  bf16* XN    = (bf16*)alloc((size_t)65536 * 192 * 2);
  bf16* QKV   = (bf16*)alloc((size_t)65536 * 576 * 2);
  bf16* QB    = (bf16*)alloc((size_t)1536 * 256 * 32 * 2);
  bf16* KB    = (bf16*)alloc((size_t)1536 * 256 * 32 * 2);
  bf16* VB    = (bf16*)alloc((size_t)1536 * 32 * 256 * 2);
  bf16* OB    = (bf16*)alloc((size_t)65536 * 192 * 2);
  float* ATTN = (float*)alloc((size_t)65536 * 180 * 4);
  bf16* Y1    = (bf16*)alloc((size_t)65536 * 64 * 2);
  float* Y2   = (float*)alloc((size_t)65536 * 180 * 4);
  float* POOL = (float*)alloc(720 * 4);
  float* AVEC = (float*)alloc(720 * 4);
  float* XRES = (float*)alloc((size_t)65536 * 180 * 4);
  bf16* XN2   = (bf16*)alloc((size_t)65536 * 192 * 2);
  bf16* WQKV  = (bf16*)alloc((size_t)576 * 192 * 2);
  bf16* WPROJ = (bf16*)alloc((size_t)192 * 192 * 2);
  bf16* WC1   = (bf16*)alloc((size_t)64 * 1728 * 2);
  bf16* WC2   = (bf16*)alloc((size_t)192 * 576 * 2);
  bf16* WF1   = (bf16*)alloc((size_t)768 * 192 * 2);
  bf16* WF2   = (bf16*)alloc((size_t)192 * 768 * 2);
  // fc1 activation (65536*768*2 B) exactly reuses dead QKV(+QB) region
  bf16* H1 = QKV;

  auto nblk = [](size_t total) { return (unsigned)((total + 255) / 256); };
  pack_w<<<nblk((size_t)576 * 192), 256, 0, stream>>>(qkvw, WQKV, 540, 576, 180, 192, 1);
  pack_w<<<nblk((size_t)192 * 192), 256, 0, stream>>>(prw, WPROJ, 180, 192, 180, 192, 1);
  pack_w<<<nblk((size_t)64 * 1728), 256, 0, stream>>>(c1w, WC1, 60, 64, 180, 192, 9);
  pack_w<<<nblk((size_t)192 * 576), 256, 0, stream>>>(c2w, WC2, 180, 192, 60, 64, 9);
  pack_w<<<nblk((size_t)768 * 192), 256, 0, stream>>>(f1w, WF1, 720, 768, 180, 192, 1);
  pack_w<<<nblk((size_t)192 * 768), 256, 0, stream>>>(f2w, WF2, 180, 192, 720, 768, 1);

  // LN1 -> window-ordered bf16 activations (padded 180->192)
  ln_kernel<<<8192, 256, 0, stream>>>(x, n1g, n1b, XN, 1);

  // QKV projection (bf16 out, padded stride 576)
  gemm_kernel<false, false><<<dim3(512, 9), 256, 0, stream>>>(
      XN, WQKV, qkvb, nullptr, nullptr, QKV, 540, 576, 192, 0, 0);
  repack_qkv<<<(65536 * 192) / 256, 256, 0, stream>>>(QKV, QB, KB, VB);

  // Window attention (zero O first so pad columns are exact zeros)
  hipMemsetAsync(OB, 0, (size_t)65536 * 192 * 2, stream);
  attn_kernel<<<dim3(16, 6, 256), 256, 0, stream>>>(QB, KB, VB, rpi, rpb, OB);
  gemm_kernel<false, false><<<dim3(512, 3), 256, 0, stream>>>(
      OB, WPROJ, prb, nullptr, ATTN, nullptr, 180, 192, 192, 0, 0);

  // CAB conv branch: conv3x3(GELU) -> conv3x3 -> channel attention
  gemm_kernel<true, true><<<dim3(512, 1), 256, 0, stream>>>(
      XN, WC1, c1b, nullptr, nullptr, Y1, 60, 64, 1728, 192, 1);
  gemm_kernel<true, false><<<dim3(512, 3), 256, 0, stream>>>(
      Y1, WC2, c2b, nullptr, Y2, nullptr, 180, 192, 576, 64, 0);
  pool_kernel<<<720, 256, 0, stream>>>(Y2, POOL);
  ca_kernel<<<1, 256, 0, stream>>>(POOL, ca1w, ca1b, ca2w, ca2b, AVEC);

  // x_res = shortcut + attn(window-reversed) + conv_x * 0.01
  combine_kernel<<<(65536 * 180 + 255) / 256, 256, 0, stream>>>(x, ATTN, Y2, AVEC, XRES);

  // MLP with fused residual into d_out
  ln_kernel<<<8192, 256, 0, stream>>>(XRES, n2g, n2b, XN2, 0);
  gemm_kernel<false, false><<<dim3(512, 12), 256, 0, stream>>>(
      XN2, WF1, f1b, nullptr, nullptr, H1, 720, 768, 192, 0, 1);
  gemm_kernel<false, false><<<dim3(512, 3), 256, 0, stream>>>(
      H1, WF2, f2b, XRES, out, nullptr, 180, 192, 768, 0, 0);
}
